// MultiHeadBlock_6167573037565
// MI455X (gfx1250) — matrix operand order repaired and checked by reading, not compiled, not run on hardware
//
#include <hip/hip_runtime.h>
#include <hip/hip_bf16.h>

// ---------------------------------------------------------------------------
// Problem constants (reference: B=4, S=2048, D=512, H=8, dh=64)
// ---------------------------------------------------------------------------
#define BB     4
#define SS     2048
#define DD     512
#define HH     8
#define DHH    64
#define EPS_LN 1e-5f

typedef __attribute__((ext_vector_type(16))) __bf16 v16bf;
typedef __attribute__((ext_vector_type(8)))  __bf16 v8bf;
typedef __attribute__((ext_vector_type(8)))  float  v8f;

// ---------------------------------------------------------------------------
// Async global->LDS copy (ASYNCcnt path).  Builtin signature (from hipcc
// diagnostics): param0 = AS1 (global) pointer to `int __attribute__((
// vector_size(16)))`, param1 = LDS pointer, params 2/3 = imm offset / cpol.
// Falls back to a synchronous 16-byte VGPR bounce if builtins are missing.
// ---------------------------------------------------------------------------
#if defined(__has_builtin)
#if __has_builtin(__builtin_amdgcn_global_load_async_to_lds_b128)
#define HAVE_ASYNC_LDS 1
#endif
#endif
#ifndef HAVE_ASYNC_LDS
#define HAVE_ASYNC_LDS 0
#endif

typedef int v4i_vs __attribute__((vector_size(16)));
typedef __attribute__((address_space(1))) v4i_vs* as1_v4i_ptr;
typedef __attribute__((address_space(3))) v4i_vs* as3_v4i_ptr;

__device__ __forceinline__ void async_cp16(void* lds, const void* g) {
#if HAVE_ASYNC_LDS
  __builtin_amdgcn_global_load_async_to_lds_b128(
      (as1_v4i_ptr)(unsigned long long)g,
      // flat LDS address truncated to 32 bits == LDS byte offset (ISA 10.2)
      (as3_v4i_ptr)(unsigned)(unsigned long long)lds,
      0, 0);
#else
  *(v8bf*)lds = *(const v8bf*)g;
#endif
}

#if HAVE_ASYNC_LDS
#if __has_builtin(__builtin_amdgcn_s_wait_asynccnt)
#define ASYNC_WAIT(n) __builtin_amdgcn_s_wait_asynccnt(n)
#else
#define ASYNC_WAIT(n) asm volatile("s_wait_asynccnt " #n ::: "memory")
#endif
#else
#define ASYNC_WAIT(n) ((void)0)
#endif

// ---------------------------------------------------------------------------
// WMMA helpers.  CDNA5 VGPR layouts (CDNA5 ISA §7.12.2):
//   A (16x32 bf16): lane l holds row m=l%16; elems i=0..7 -> k=8*half+i,
//                   i=8..15 -> k=16+8*half+(i-8)   (half = l/16)
//   B (32x16 bf16): lane l holds col n=l%16; same k order as A
//   C/D (16x16 f32): lane l holds col n=l%16; VGPR j -> row m = j + 8*half
// ---------------------------------------------------------------------------
__device__ __forceinline__ v16bf load_a_frag(const __bf16* rowp, int half) {
  const v8bf lo = *(const v8bf*)(rowp + 8 * half);
  const v8bf hi = *(const v8bf*)(rowp + 16 + 8 * half);
  v16bf a;
#pragma unroll
  for (int i = 0; i < 8; ++i) { a[i] = lo[i]; a[i + 8] = hi[i]; }
  return a;
}

__device__ __forceinline__ v16bf load_b_frag(const __bf16* rowp, int half) {
  const v8bf lo = *(const v8bf*)(rowp + 8 * half);
  const v8bf hi = *(const v8bf*)(rowp + 16 + 8 * half);
  v16bf b;
#pragma unroll
  for (int i = 0; i < 8; ++i) { b[i] = lo[i]; b[i + 8] = hi[i]; }
  return b;
}

__device__ __forceinline__ v8f wmma_bf16(v16bf a, v16bf b, v8f c) {
  return __builtin_amdgcn_wmma_f32_16x16x32_bf16(
      false, a, false, b, (short)0, c, false, false);
}

// 16-lane (half-wave) reductions — wave32: xor masks < 16 stay inside halves.
__device__ __forceinline__ float red_max16(float v) {
#pragma unroll
  for (int m = 1; m < 16; m <<= 1) v = fmaxf(v, __shfl_xor(v, m, 32));
  return v;
}
__device__ __forceinline__ float red_sum16(float v) {
#pragma unroll
  for (int m = 1; m < 16; m <<= 1) v += __shfl_xor(v, m, 32);
  return v;
}

// ---------------------------------------------------------------------------
// Kernel 1: QKV projection.  [8192 x 512] @ [512 x 1536] + bias.
// 64x64 tile per WG (8 waves, 4(M) x 2(N)), BK=32, bf16 WMMA, fp32 accum.
// Writes Q,K as [B,H,S,dh] bf16 and V transposed [B,H,dh,S] bf16.
// ---------------------------------------------------------------------------
#define LDS_STRIDE 40   // 32 bf16 payload + pad; 80 B keeps 16B alignment

__global__ __launch_bounds__(256) void qkv_kernel(
    const float* __restrict__ x, const float* __restrict__ Wqkv,
    const float* __restrict__ bqkv, __bf16* __restrict__ Q,
    __bf16* __restrict__ K, __bf16* __restrict__ Vt) {
  __shared__ __align__(16) __bf16 As[64 * LDS_STRIDE];   // [m][k]
  __shared__ __align__(16) __bf16 Bs[64 * LDS_STRIDE];   // [n][k] (transposed)

  const int t = threadIdx.x;
  const int m0 = blockIdx.x * 64;
  const int n0 = blockIdx.y * 64;
  const int w = t >> 5, l = t & 31;
  const int half = l >> 4, ln = l & 15;
  const int m_off = (w & 3) * 16;
  const int n_off = (w >> 2) * 32;

  v8f acc[2] = {};

  for (int k0 = 0; k0 < DD; k0 += 32) {
    __syncthreads();
#pragma unroll
    for (int r = 0; r < 8; ++r) {
      const int idx = t + 256 * r;                       // 2048 elems each
      const int mm = idx >> 5, kk = idx & 31;            // A: 64x32
      As[mm * LDS_STRIDE + kk] = (__bf16)x[(size_t)(m0 + mm) * DD + k0 + kk];
      const int nn = idx & 63, k2 = idx >> 6;            // B: 32x64 -> [n][k]
      Bs[nn * LDS_STRIDE + k2] =
          (__bf16)Wqkv[(size_t)(k0 + k2) * (3 * DD) + n0 + nn];
    }
    __syncthreads();

    const v16bf a = load_a_frag(&As[(m_off + ln) * LDS_STRIDE], half);
#pragma unroll
    for (int sub = 0; sub < 2; ++sub) {
      const v16bf b =
          load_b_frag(&Bs[(n_off + 16 * sub + ln) * LDS_STRIDE], half);
      acc[sub] = wmma_bf16(a, b, acc[sub]);
    }
  }

  // Epilogue.  `sec` is uniform per WG (64-wide N tile never crosses a 512
  // boundary) -> scalar branch, no exec-mask churn.
  const int sec = n0 >> 9;                               // 0=Q 1=K 2=V
  const int b_i = m0 >> 11;                              // batch (uniform)
  const int s_base = (m0 & (SS - 1)) + m_off + 8 * half; // this lane's 1st row
#pragma unroll
  for (int sub = 0; sub < 2; ++sub) {
    const int c = (n0 & (DD - 1)) + n_off + 16 * sub + ln; // col in section
    const float bias = bqkv[(sec << 9) + c];
    const int h = c >> 6, d = c & (DHH - 1);
    const size_t bh = (size_t)b_i * HH + h;
    if (sec == 2) {
      // V^T: contiguous in s per lane -> one 16B vector store
      v8bf vv;
#pragma unroll
      for (int j = 0; j < 8; ++j) vv[j] = (__bf16)(acc[sub][j] + bias);
      *(v8bf*)(Vt + (bh * DHH + d) * SS + s_base) = vv;
    } else {
      __bf16* p = (sec == 0 ? Q : K) + (bh * SS + s_base) * DHH + d;
#pragma unroll
      for (int j = 0; j < 8; ++j)
        p[(size_t)j * DHH] = (__bf16)(acc[sub][j] + bias);
    }
  }
}

// ---------------------------------------------------------------------------
// Kernel 2: flash attention (unscaled).  WG = 8 waves x 16 q rows of one
// (b,h).  KV streamed in chunks of 32 through LDS, staged ONCE per WG with
// double-buffered GLOBAL_LOAD_ASYNC_TO_LDS_B128 (ASYNCcnt), so the 8 waves
// share one copy instead of 8 redundant global streams.
// ---------------------------------------------------------------------------
#define KSTR 72   // K chunk row: 64 bf16 payload + pad (144 B, 16B aligned)
#define VSTR 40   // V chunk row: 32 bf16 payload + pad ( 80 B, 16B aligned)

__global__ __launch_bounds__(256) void attn_kernel(
    const __bf16* __restrict__ Q, const __bf16* __restrict__ K,
    const __bf16* __restrict__ Vt, const int* __restrict__ mask,
    float* __restrict__ att) {
  __shared__ __align__(16) __bf16 Ks[2][32 * KSTR];          // [kv][d]
  __shared__ __align__(16) __bf16 Vs[2][64 * VSTR];          // [d][kv]
  __shared__ __align__(16) __bf16 Pl[8 * 16 * LDS_STRIDE];   // per-wave P

  const int t = threadIdx.x, w = t >> 5, l = t & 31;
  const int half = l >> 4, ln = l & 15;
  const int bh = blockIdx.y;                   // 0..B*H-1
  const int b = bh >> 3, h = bh & 7;
  const int q0 = blockIdx.x * 128 + w * 16;

  const __bf16* Kbase = K + (size_t)bh * SS * DHH;
  const __bf16* Vbase = Vt + (size_t)bh * DHH * SS;
  const int* mbase = mask + b * SS;

  // Per-thread staging slots: K chunk 32x64 bf16 (4 KB), V chunk 64x32 (4 KB);
  // 256 threads x 16 B each.
  const int kr = t >> 3, kc = (t & 7) * 8;     // K: row kv, col d
  const int vr = t >> 2, vc = (t & 3) * 8;     // V: row d,  col kv

  // Q fragments for this wave's 16 rows (dh=64 -> two 16x32 A frags)
  const __bf16* Qp = Q + ((size_t)bh * SS + q0 + ln) * DHH;
  const v16bf aQ0 = load_a_frag(Qp, half);
  const v16bf aQ1 = load_a_frag(Qp + 32, half);

  float m_run[8], l_run[8];
#pragma unroll
  for (int j = 0; j < 8; ++j) { m_run[j] = -1e30f; l_run[j] = 0.f; }
  v8f o[4] = {};
  __bf16* Pw = &Pl[w * 16 * LDS_STRIDE];

  // Prologue: stage chunk 0 into buffer 0.
  async_cp16(&Ks[0][kr * KSTR + kc], Kbase + (size_t)kr * DHH + kc);
  async_cp16(&Vs[0][vr * VSTR + vc], Vbase + (size_t)vr * SS + vc);

  for (int kv0 = 0; kv0 < SS; kv0 += 32) {
    const int buf = (kv0 >> 5) & 1;
    if (kv0 + 32 < SS) {     // stage next chunk, then retire current (<=2 left)
      const int nxt = kv0 + 32;
      async_cp16(&Ks[buf ^ 1][kr * KSTR + kc],
                 Kbase + (size_t)(nxt + kr) * DHH + kc);
      async_cp16(&Vs[buf ^ 1][vr * VSTR + vc],
                 Vbase + (size_t)vr * SS + nxt + vc);
      ASYNC_WAIT(2);
    } else {
      ASYNC_WAIT(0);
    }
    __syncthreads();
    const __bf16* Kb = &Ks[buf][0];
    const __bf16* Vb = &Vs[buf][0];

    // scores: S(16x32) = Q(16x64) @ K^T(64x32), two column halves
    v8f s0 = {}, s1 = {};
    s0 = wmma_bf16(aQ0, load_b_frag(Kb + ln * KSTR, half), s0);
    s0 = wmma_bf16(aQ1, load_b_frag(Kb + ln * KSTR + 32, half), s0);
    s1 = wmma_bf16(aQ0, load_b_frag(Kb + (16 + ln) * KSTR, half), s1);
    s1 = wmma_bf16(aQ1, load_b_frag(Kb + (16 + ln) * KSTR + 32, half), s1);

    // key mask (lane = score column)
    const int mk0 = mbase[kv0 + ln];
    const int mk1 = mbase[kv0 + 16 + ln];
#pragma unroll
    for (int j = 0; j < 8; ++j) {
      s0[j] = mk0 ? s0[j] : -1e30f;
      s1[j] = mk1 ? s1[j] : -1e30f;
    }

    // online softmax per row (rows live across 16-lane halves)
#pragma unroll
    for (int j = 0; j < 8; ++j) {
      const float mx = red_max16(fmaxf(s0[j], s1[j]));
      const float mn = fmaxf(m_run[j], mx);
      const float corr = __expf(m_run[j] - mn);
      const float p0 = __expf(s0[j] - mn);
      const float p1 = __expf(s1[j] - mn);
      const float rs = red_sum16(p0 + p1);
      l_run[j] = l_run[j] * corr + rs;
      m_run[j] = mn;
#pragma unroll
      for (int f = 0; f < 4; ++f) o[f][j] *= corr;
      // C-layout -> LDS (row j+8*half, cols ln / ln+16)
      Pw[(j + 8 * half) * LDS_STRIDE + ln]      = (__bf16)p0;
      Pw[(j + 8 * half) * LDS_STRIDE + 16 + ln] = (__bf16)p1;
    }
    asm volatile("s_wait_dscnt 0" ::: "memory");  // wave-local LDS transpose

    // O(16x64) += P(16x32) @ V(32x64)
    const v16bf aP = load_a_frag(&Pw[ln * LDS_STRIDE], half);
#pragma unroll
    for (int f = 0; f < 4; ++f)
      o[f] = wmma_bf16(aP, load_b_frag(Vb + (16 * f + ln) * VSTR, half), o[f]);

    __syncthreads();   // all waves done with buf before it is restaged
  }

  // normalize and store att[b, s, h*64 + d] fp32
  float* arow = att + (size_t)b * SS * DD;
#pragma unroll
  for (int f = 0; f < 4; ++f) {
    const int col = h * DHH + 16 * f + ln;
#pragma unroll
    for (int j = 0; j < 8; ++j) {
      const int q = q0 + j + 8 * half;
      const float inv = (l_run[j] > 0.f) ? (1.f / l_run[j]) : 0.f;
      arow[(size_t)q * DD + col] = o[f][j] * inv;
    }
  }
}

// ---------------------------------------------------------------------------
// Kernel 3: out = att @ W_o + b_o;  x = out + att;  LayerNorm(x).
// One WG owns 16 rows x all 512 cols (8 waves x 64 cols) so the LN
// reduction is fused via shuffles + ds_add_f32 shared atomics.
// ---------------------------------------------------------------------------
__global__ __launch_bounds__(256) void proj_ln_kernel(
    const float* __restrict__ att, const float* __restrict__ Wo,
    const float* __restrict__ bo, const float* __restrict__ gamma,
    const float* __restrict__ beta, float* __restrict__ out) {
  __shared__ __align__(16) __bf16 As[16 * LDS_STRIDE];      // [m][k]
  __shared__ __align__(16) __bf16 Bs[DD * LDS_STRIDE];      // [n][k]
  __shared__ float s_sum[16], s_sq[16];

  const int t = threadIdx.x, w = t >> 5, l = t & 31;
  const int half = l >> 4, ln = l & 15;
  const int m0 = blockIdx.x * 16;

  if (t < 16) { s_sum[t] = 0.f; s_sq[t] = 0.f; }

  v8f acc[4] = {};
  for (int k0 = 0; k0 < DD; k0 += 32) {
    __syncthreads();
#pragma unroll
    for (int r = 0; r < 2; ++r) {                 // A tile: 16x32
      const int id = t + 256 * r;
      const int mm = id >> 5, kk = id & 31;
      As[mm * LDS_STRIDE + kk] =
          (__bf16)att[(size_t)(m0 + mm) * DD + k0 + kk];
    }
#pragma unroll
    for (int r = 0; r < 64; ++r) {                // B tile: 32x512 -> [n][k]
      const int id = t + 256 * r;
      const int nn = id & (DD - 1), kk = id >> 9;
      Bs[nn * LDS_STRIDE + kk] = (__bf16)Wo[(size_t)(k0 + kk) * DD + nn];
    }
    __syncthreads();

    const v16bf a = load_a_frag(&As[ln * LDS_STRIDE], half);
#pragma unroll
    for (int f = 0; f < 4; ++f) {
      const v16bf bb =
          load_b_frag(&Bs[(w * 64 + 16 * f + ln) * LDS_STRIDE], half);
      acc[f] = wmma_bf16(a, bb, acc[f]);
    }
  }

  // x = GEMM + bias + residual(att)
  float xv[4][8];
#pragma unroll
  for (int f = 0; f < 4; ++f) {
    const int col = w * 64 + 16 * f + ln;
    const float bias = bo[col];
#pragma unroll
    for (int j = 0; j < 8; ++j) {
      const int row = m0 + j + 8 * half;
      xv[f][j] = acc[f][j] + bias + att[(size_t)row * DD + col];
    }
  }

  // LayerNorm reductions: 4 cols/lane -> 16-lane shuffle -> ds_add_f32
#pragma unroll
  for (int j = 0; j < 8; ++j) {
    float s = 0.f, q = 0.f;
#pragma unroll
    for (int f = 0; f < 4; ++f) { s += xv[f][j]; q += xv[f][j] * xv[f][j]; }
    s = red_sum16(s);
    q = red_sum16(q);
    if (ln == 0) {
      atomicAdd(&s_sum[j + 8 * half], s);
      atomicAdd(&s_sq[j + 8 * half], q);
    }
  }
  __syncthreads();

#pragma unroll
  for (int f = 0; f < 4; ++f) {
    const int col = w * 64 + 16 * f + ln;
    const float g = gamma[col], bt = beta[col];
#pragma unroll
    for (int j = 0; j < 8; ++j) {
      const int r8 = j + 8 * half;
      const float mean = s_sum[r8] * (1.f / (float)DD);
      const float var = s_sq[r8] * (1.f / (float)DD) - mean * mean;
      const float rstd = rsqrtf(var + EPS_LN);
      out[(size_t)(m0 + r8) * DD + col] = (xv[f][j] - mean) * rstd * g + bt;
    }
  }
}

// ---------------------------------------------------------------------------
// Host-side launcher.
// Workspace layout (bytes): Q bf16 8MB | K bf16 8MB | V^T bf16 8MB | att f32 16MB
// ---------------------------------------------------------------------------
extern "C" void kernel_launch(void* const* d_in, const int* in_sizes, int n_in,
                              void* d_out, int out_size, void* d_ws,
                              size_t ws_size, hipStream_t stream) {
  (void)in_sizes; (void)n_in; (void)out_size; (void)ws_size;

  const float* x     = (const float*)d_in[0];
  const int*   mask  = (const int*)d_in[1];
  const float* Wqkv  = (const float*)d_in[2];
  const float* bqkv  = (const float*)d_in[3];
  const float* Wo    = (const float*)d_in[4];
  const float* bo    = (const float*)d_in[5];
  const float* gamma = (const float*)d_in[6];
  const float* beta  = (const float*)d_in[7];
  float* out = (float*)d_out;

  const size_t qkvElems = (size_t)BB * HH * SS * DHH;      // 4M elems
  char* ws = (char*)d_ws;
  __bf16* Q  = (__bf16*)(ws);
  __bf16* K  = (__bf16*)(ws + qkvElems * 2);
  __bf16* Vt = (__bf16*)(ws + qkvElems * 4);
  float* att = (float*)(ws + qkvElems * 6);

  // 1) QKV projection: M=8192, N=1536 in 64x64 tiles.
  dim3 g1((BB * SS) / 64, (3 * DD) / 64);
  qkv_kernel<<<g1, 256, 0, stream>>>(x, Wqkv, bqkv, Q, K, Vt);

  // 2) Flash attention: 128 q-rows per WG, one (b,h) per grid-y.
  dim3 g2(SS / 128, BB * HH);
  attn_kernel<<<g2, 256, 0, stream>>>(Q, K, Vt, mask, att);

  // 3) Output projection + residual + LayerNorm: 16 rows per WG.
  dim3 g3((BB * SS) / 16);
  proj_ln_kernel<<<g3, 256, 0, stream>>>(att, Wo, bo, gamma, beta, out);
}
